// GcnEncoder_43997644980265
// MI455X (gfx1250) — matrix operand order repaired and checked by reading, not compiled, not run on hardware
//
#include <hip/hip_runtime.h>
#include <hip/hip_bf16.h>

#define HID 128
#define NLAYERS 3

typedef __attribute__((ext_vector_type(16))) __bf16 v16bf;
typedef __attribute__((ext_vector_type(8)))  float  v8f;

// ---------------------------------------------------------------------------
// 1) Embedding gather + concat: x[n] = [lane(16) | type(32) | length(16) | node(64)]
// ---------------------------------------------------------------------------
__global__ void embed_kernel(const int* __restrict__ node_f,
                             const int* __restrict__ type_f,
                             const int* __restrict__ len_f,
                             const int* __restrict__ lane_f,
                             const float* __restrict__ node_emb,
                             const float* __restrict__ type_emb,
                             const float* __restrict__ len_emb,
                             const float* __restrict__ lane_emb,
                             float* __restrict__ X) {
  const int n = blockIdx.x;
  const int j = threadIdx.x;        // 0..127
  float v;
  if (j < 16)       v = lane_emb[lane_f[n] * 16 + j];
  else if (j < 48)  v = type_emb[type_f[n] * 32 + (j - 16)];
  else if (j < 64)  v = len_emb [len_f [n] * 16 + (j - 48)];
  else              v = node_emb[(size_t)node_f[n] * 64 + (j - 64)];
  X[(size_t)n * HID + j] = v;
}

// ---------------------------------------------------------------------------
// 2) Split W (fp32) into bf16 hi/lo pair, stored TRANSPOSED ([col][k]) so the
//    WMMA B-fragment (16 contiguous k per lane) is a single 32B vector load.
// ---------------------------------------------------------------------------
__global__ void splitW_kernel(const float* __restrict__ W,
                              __bf16* __restrict__ WhiT,
                              __bf16* __restrict__ WloT) {
  const int i   = blockIdx.x * blockDim.x + threadIdx.x;   // 0..16383
  const int k   = i >> 7;          // row of W
  const int col = i & 127;         // col of W
  const float f = W[i];
  const __bf16 h = (__bf16)f;
  WhiT[col * HID + k] = h;
  WloT[col * HID + k] = (__bf16)(f - (float)h);
}

// ---------------------------------------------------------------------------
// 3) support = X @ W via V_WMMA_F32_16X16X32_BF16, bf16x3 accuracy recovery.
//    One wave per 16x16 output tile; 8 waves/block -> 16 rows x 128 cols.
//    A and B use contiguous k order, k = 16*half + i (wave32); all loads vectorized.
// ---------------------------------------------------------------------------
__global__ void __launch_bounds__(256)
gemm_wmma_kernel(const float* __restrict__ X,
                 const __bf16* __restrict__ WhiT,
                 const __bf16* __restrict__ WloT,
                 float* __restrict__ S) {
  const int wave    = threadIdx.x >> 5;     // 0..7 -> 16-col tile
  const int lane    = threadIdx.x & 31;
  const int rowTile = blockIdx.x * 16;      // N = 100000 = 6250 * 16 exactly
  const int colTile = wave * 16;
  const int half    = lane >> 4;            // 0: lanes 0-15, 1: lanes 16-31
  const int rc      = lane & 15;            // A: row index; B: col index

  const float*  xrow = X + (size_t)(rowTile + rc) * HID;
  const __bf16* whc  = WhiT + (size_t)(colTile + rc) * HID;
  const __bf16* wlc  = WloT + (size_t)(colTile + rc) * HID;

  v8f acc = {};
  #pragma unroll
  for (int kb = 0; kb < HID; kb += 32) {
    // A fragment (16x32 bf16): lane rc holds row rc; element i ->
    // k = kb + 16*half + i  == two contiguous 8-float runs
    const v8f x0 = *(const v8f*)(xrow + kb + half * 16);        // 32B aligned
    const v8f x1 = *(const v8f*)(xrow + kb + 8 + half * 16);   // 32B aligned
    v16bf ahi, alo;
    #pragma unroll
    for (int i = 0; i < 8; ++i) {
      float f = x0[i];
      __bf16 h = (__bf16)f;
      ahi[i] = h;
      alo[i] = (__bf16)(f - (float)h);
      f = x1[i];
      h = (__bf16)f;
      ahi[i + 8] = h;
      alo[i + 8] = (__bf16)(f - (float)h);
    }
    // B fragment (32x16 bf16): lane rc holds column rc; element i ->
    // k = kb + 16*half + i  == 16 contiguous bf16 in transposed W (32B aligned)
    const v16bf bhi = *(const v16bf*)(whc + kb + half * 16);
    const v16bf blo = *(const v16bf*)(wlc + kb + half * 16);

    // fp32-accurate product: lo*hi + hi*lo + hi*hi, f32 accumulate
    acc = __builtin_amdgcn_wmma_f32_16x16x32_bf16(false, alo, false, bhi,
                                                  (short)0, acc, false, false);
    acc = __builtin_amdgcn_wmma_f32_16x16x32_bf16(false, ahi, false, blo,
                                                  (short)0, acc, false, false);
    acc = __builtin_amdgcn_wmma_f32_16x16x32_bf16(false, ahi, false, bhi,
                                                  (short)0, acc, false, false);
  }
  // C/D layout: VGPR r -> row = rowTile + r + 8*half, col = colTile + rc
  const int rBase = rowTile + half * 8;
  const int col   = colTile + rc;
  #pragma unroll
  for (int r = 0; r < 8; ++r)
    S[(size_t)(rBase + r) * HID + col] = acc[r];
}

// ---------------------------------------------------------------------------
// 4) Y[n][j] = b[j]  (fold bias into the scatter target's initial value)
// ---------------------------------------------------------------------------
__global__ void initb_kernel(const float* __restrict__ b, float* __restrict__ Y) {
  Y[(size_t)blockIdx.x * HID + threadIdx.x] = b[threadIdx.x];
}

// ---------------------------------------------------------------------------
// 5) COO SpMM scatter: Y[dst] += val * S[src]. One wave per edge, one float4
//    per lane (128 floats = 32 lanes x 4). Native f32 atomics in L2.
// ---------------------------------------------------------------------------
__global__ void __launch_bounds__(256)
scatter_kernel(const float4* __restrict__ S4,
               const int* __restrict__ src,
               const int* __restrict__ dst,
               const float* __restrict__ val,
               float* __restrict__ Y, int E) {
  const int t = blockIdx.x * blockDim.x + threadIdx.x;
  const int e = t >> 5;
  if (e >= E) return;
  const int g = t & 31;
  const int s = src[e];
  const int d = dst[e];
  const float v = val[e];
  const float4 m = S4[(size_t)s * 32 + g];
  float* yp = Y + (size_t)d * HID + g * 4;
  atomicAdd(yp + 0, m.x * v);
  atomicAdd(yp + 1, m.y * v);
  atomicAdd(yp + 2, m.z * v);
  atomicAdd(yp + 3, m.w * v);
}

// ---------------------------------------------------------------------------
extern "C" void kernel_launch(void* const* d_in, const int* in_sizes, int n_in,
                              void* d_out, int out_size, void* d_ws, size_t ws_size,
                              hipStream_t stream) {
  const int*   node_f   = (const int*)  d_in[0];
  const int*   type_f   = (const int*)  d_in[1];
  const int*   len_f    = (const int*)  d_in[2];
  const int*   lane_f   = (const int*)  d_in[3];
  const int*   adj_src  = (const int*)  d_in[4];
  const int*   adj_dst  = (const int*)  d_in[5];
  const float* adj_val  = (const float*)d_in[6];
  const float* node_emb = (const float*)d_in[7];
  const float* type_emb = (const float*)d_in[8];
  const float* len_emb  = (const float*)d_in[9];
  const float* lane_emb = (const float*)d_in[10];
  const float* W        = (const float*)d_in[11];
  const float* b        = (const float*)d_in[12];

  const int N = in_sizes[0];   // 100000 (multiple of 16)
  const int E = in_sizes[4];   // 640000

  float*  X    = (float*)d_out;                        // node features, ping-pong in place
  float*  S    = (float*)d_ws;                         // support = X @ W   (N*HID floats)
  __bf16* WhiT = (__bf16*)((char*)d_ws + (size_t)N * HID * sizeof(float));
  __bf16* WloT = WhiT + HID * HID;

  // build x from embeddings
  embed_kernel<<<N, HID, 0, stream>>>(node_f, type_f, len_f, lane_f,
                                      node_emb, type_emb, len_emb, lane_emb, X);
  // bf16 hi/lo split of W, transposed (once per call; deterministic)
  splitW_kernel<<<(HID * HID) / 256, 256, 0, stream>>>(W, WhiT, WloT);

  const int gemmGrid    = N / 16;                       // 6250
  const int scatterGrid = (E * 32 + 255) / 256;         // 80000

  for (int layer = 0; layer < NLAYERS; ++layer) {
    gemm_wmma_kernel<<<gemmGrid, 256, 0, stream>>>(X, WhiT, WloT, S);
    initb_kernel<<<N, HID, 0, stream>>>(b, X);          // X := b (broadcast)
    scatter_kernel<<<scatterGrid, 256, 0, stream>>>((const float4*)S,
                                                    adj_src, adj_dst, adj_val, X, E);
  }
}
